// Encoder_25486335935119
// MI455X (gfx1250) — compile-verified, matrix operand order checked by reading, not run on hardware
//
#include <hip/hip_runtime.h>
#include <hip/hip_bf16.h>

#define BATCH   128
#define SEQ     256
#define VOCAB   1024
#define HIDDEN  512
#define NGATE4  2048        // 4*HIDDEN
#define EOS_ID  1
#define NWG     32          // persistent workgroups (16 hidden units each)
#define TPB     256         // 8 wave32
#define LDS_STRIDE 520      // padded K stride (bf16 elems) -> conflict-free ds_load_b128

typedef __attribute__((ext_vector_type(16))) __bf16 v16bf;
typedef __attribute__((ext_vector_type(8)))  __bf16 v8bf;
typedef __attribute__((ext_vector_type(8)))  float  v8f;

union V16U { v16bf v; v8bf h[2]; };

// ---------------------------------------------------------------------------
// Kernel 1: one-hot -> token ids (wave-parallel argmax over VOCAB)
// ---------------------------------------------------------------------------
__global__ void tok_kernel(const float* __restrict__ onehot, int* __restrict__ tokens) {
    const int lane = threadIdx.x & 31;
    const int wave = threadIdx.x >> 5;
    const int gw   = blockIdx.x * (TPB / 32) + wave;   // global wave id
    #pragma unroll
    for (int rr = 0; rr < 4; ++rr) {
        const int row = gw * 4 + rr;                   // (b*SEQ + t)
        if (row >= BATCH * SEQ) break;
        const float* p = onehot + (size_t)row * VOCAB;
        int tok = 0;
        for (int i = lane; i < VOCAB; i += 32)
            if (p[i] > 0.5f) tok = i;
        for (int off = 16; off > 0; off >>= 1)
            tok = max(tok, __shfl_xor(tok, off));
        if (lane == 0) tokens[row] = tok;
    }
}

// ---------------------------------------------------------------------------
// Kernel 2: zero h(0) exchange buffer + barrier counter (every launch)
// ---------------------------------------------------------------------------
__global__ void init_kernel(__bf16* __restrict__ h0, unsigned* __restrict__ cnt) {
    const int i = blockIdx.x * blockDim.x + threadIdx.x;
    if (i < BATCH * HIDDEN) h0[i] = (__bf16)0.0f;
    if (i == 0) *cnt = 0u;
}

// ---------------------------------------------------------------------------
// Kernel 3: persistent LSTM scan. WG g owns hidden units [16g, 16g+16).
// Wave w owns batches [16w, 16w+16) of that slice. c/h for the tile live in
// registers; h is exchanged via ping-pong bf16 buffers + grid barrier.
// ---------------------------------------------------------------------------
__launch_bounds__(TPB, 1)
__global__ void lstm_kernel(const float* __restrict__ Wi,
                            const float* __restrict__ Wh,
                            const float* __restrict__ bias_g,
                            const int*   __restrict__ tokens,
                            __bf16*      __restrict__ hbuf0,
                            __bf16*      __restrict__ hbuf1,
                            unsigned*    __restrict__ cnt,
                            float*       __restrict__ out /* c then h */) {
    __shared__ __bf16 whT[64 * LDS_STRIDE];   // 66560 B: [gate*16+n][K], bf16

    const int wg   = blockIdx.x;              // 0..31
    const int j0   = wg * 16;                 // hidden-unit base
    const int tid  = threadIdx.x;
    const int lane = tid & 31;
    const int wave = tid >> 5;

    // ---- Stage this WG's Wh slice into LDS (transposed, bf16) ----
    for (int e = tid; e < 64 * HIDDEN; e += TPB) {
        const int lc   = e >> 9;              // 0..63 = gate*16 + n
        const int k    = e & (HIDDEN - 1);
        const int gate = lc >> 4, nn = lc & 15;
        const float w  = Wh[(size_t)k * NGATE4 + gate * HIDDEN + j0 + nn];
        whT[lc * LDS_STRIDE + k] = (__bf16)w;
    }
    __syncthreads();

    const int n   = lane & 15;                // output column within 16-tile
    const int hh  = lane >> 4;                // half-wave: 0 or 1
    const int bmA = wave * 16 + n;            // A-fragment batch row (M = lane%16)
    const int bmC = wave * 16 + hh * 8;       // C-fragment batch base (+r)

    float bias[4];
    #pragma unroll
    for (int G = 0; G < 4; ++G) bias[G] = bias_g[G * HIDDEN + j0 + n];

    const v8f vzero = {0.f, 0.f, 0.f, 0.f, 0.f, 0.f, 0.f, 0.f};
    v8f cfrag = vzero;
    v8f hfrag = vzero;
    unsigned eosmask = 0u;                    // bit (hh*8+r) per batch row of this wave

    for (int t = 0; t < SEQ; ++t) {
        const __bf16* hread  = (t & 1) ? hbuf1 : hbuf0;
        __bf16*       hwrite = (t & 1) ? hbuf0 : hbuf1;

        v8f z[4];
        #pragma unroll
        for (int G = 0; G < 4; ++G) z[G] = vzero;

        // ---- z += h(t) @ Wh  via v_wmma_f32_16x16x32_bf16 ----
        for (int kk = 0; kk < HIDDEN; kk += 32) {
            V16U A;   // lane holds row bmA, K = kk + {kb..kb+7, kb+16..kb+23}, kb = hh*8
            const __bf16* ap = hread + (size_t)bmA * HIDDEN + kk + hh * 8;
            A.h[0] = *(const v8bf*)(ap);
            A.h[1] = *(const v8bf*)(ap + 16);
            #pragma unroll
            for (int G = 0; G < 4; ++G) {
                V16U B;  // lane holds column G*16+n, same K pattern, from LDS
                const __bf16* bp = &whT[(G * 16 + n) * LDS_STRIDE + kk + hh * 8];
                B.h[0] = *(const v8bf*)(bp);
                B.h[1] = *(const v8bf*)(bp + 16);
                z[G] = __builtin_amdgcn_wmma_f32_16x16x32_bf16(
                    false, A.v, false, B.v, (short)0, z[G], false, false);
            }
        }

        // ---- tokens for this lane's 8 C rows; x@Wi is a row gather ----
        int tk[8];
        #pragma unroll
        for (int r = 0; r < 8; ++r) tk[r] = tokens[(size_t)(bmC + r) * SEQ + t];

        #pragma unroll
        for (int G = 0; G < 4; ++G) {
            #pragma unroll
            for (int r = 0; r < 8; ++r) {
                z[G][r] += Wi[(size_t)tk[r] * NGATE4 + G * HIDDEN + j0 + n] + bias[G];
            }
        }

        // ---- gates + state update, frozen by OLD eos mask (reference order) ----
        #pragma unroll
        for (int r = 0; r < 8; ++r) {
            const bool frozen = (eosmask >> (hh * 8 + r)) & 1u;
            const float ig = 1.f / (1.f + __expf(-z[0][r]));
            const float fg = 1.f / (1.f + __expf(-z[1][r]));
            const float gg = tanhf(z[2][r]);
            const float og = 1.f / (1.f + __expf(-z[3][r]));
            const float nc = fg * cfrag[r] + ig * gg;
            const float nh = og * tanhf(nc);
            if (!frozen) { cfrag[r] = nc; hfrag[r] = nh; }
            hwrite[(size_t)(bmC + r) * HIDDEN + j0 + n] = (__bf16)hfrag[r];
        }

        // ---- eos |= (token == EOS_ID), merged across half-waves ----
        unsigned mybits = 0u;
        #pragma unroll
        for (int r = 0; r < 8; ++r)
            if (tk[r] == EOS_ID) mybits |= (1u << (hh * 8 + r));
        mybits |= __shfl_xor(mybits, 16);
        eosmask |= mybits;

        // ---- grid-wide step barrier (agent-scope atomic, monotonic target) ----
        __syncthreads();
        if (tid == 0) {
            __hip_atomic_fetch_add(cnt, 1u, __ATOMIC_RELEASE, __HIP_MEMORY_SCOPE_AGENT);
            const unsigned target = (unsigned)NWG * (unsigned)(t + 1);
            while (__hip_atomic_load(cnt, __ATOMIC_ACQUIRE, __HIP_MEMORY_SCOPE_AGENT) < target)
                __builtin_amdgcn_s_sleep(2);
        }
        __syncthreads();
    }

    // ---- write f32 outputs: c (exact, never left registers) then h ----
    #pragma unroll
    for (int r = 0; r < 8; ++r) {
        const size_t idx = (size_t)(bmC + r) * HIDDEN + j0 + n;
        out[idx] = cfrag[r];
        out[(size_t)BATCH * HIDDEN + idx] = hfrag[r];
    }
}

// ---------------------------------------------------------------------------
extern "C" void kernel_launch(void* const* d_in, const int* in_sizes, int n_in,
                              void* d_out, int out_size, void* d_ws, size_t ws_size,
                              hipStream_t stream) {
    const float* onehot = (const float*)d_in[0];   // (128,256,1024) f32 one-hot
    const float* Wi     = (const float*)d_in[1];   // (1024,2048) f32
    const float* Wh     = (const float*)d_in[2];   // (512,2048)  f32
    const float* b      = (const float*)d_in[3];   // (2048,)     f32
    float*       out    = (float*)d_out;           // c (128x512) then h (128x512)

    char*     ws     = (char*)d_ws;
    int*      tokens = (int*)ws;                         // 131072 B
    __bf16*   hbuf0  = (__bf16*)(ws + 131072);           // 131072 B
    __bf16*   hbuf1  = (__bf16*)(ws + 262144);           // 131072 B
    unsigned* cnt    = (unsigned*)(ws + 393216);         // 4 B

    tok_kernel <<<BATCH * SEQ / 32, TPB, 0, stream>>>(onehot, tokens);
    init_kernel<<<(BATCH * HIDDEN + TPB - 1) / TPB, TPB, 0, stream>>>(hbuf0, cnt);
    lstm_kernel<<<NWG, TPB, 0, stream>>>(Wi, Wh, b, tokens, hbuf0, hbuf1, cnt, out);
}
